// MCFModel_10728828305628
// MI455X (gfx1250) — compile-verified, matrix operand order checked by reading, not run on hardware
//
#include <hip/hip_runtime.h>
#include <math.h>

typedef __attribute__((ext_vector_type(16))) _Float16 v16h;
typedef __attribute__((ext_vector_type(8)))  _Float16 v8h;
typedef __attribute__((ext_vector_type(8)))  float    v8f;

#define KN 3072
#define KD 256
#define KH 8

// ---- WMMA lane-layout helpers (CDNA5 ISA 7.12.2, wave32) -------------------
// 16-bit A matrix 16x32: element i of v16h maps to K = i + 8*half + (i>=8?8:0)
// => per lane, two contiguous 8-element runs: [8h..8h+7] and [16+8h..16+8h+7].
// B (32x16): lane (lane&15) = column n, same K mapping => store B transposed
// in LDS (Bt[n][k], stride 32 f16) so each fragment is two 16B ds_load_b128.
// C/D f32 16x16: element r -> m = r + 8*half, n = lane&15.
__device__ __forceinline__ float lrelu02(float x) { return x > 0.f ? x : 0.2f * x; }
__device__ __forceinline__ float sigmoidf_(float x) { return 1.f / (1.f + __expf(-x)); }

__device__ __forceinline__ v16h joinv(v8h a, v8h b) {
    return __builtin_shufflevector(a, b, 0,1,2,3,4,5,6,7,8,9,10,11,12,13,14,15);
}
__device__ __forceinline__ v16h pack_a(float4 a0, float4 a1, float4 a2, float4 a3) {
    v16h af;
    af[0]=(_Float16)a0.x; af[1]=(_Float16)a0.y; af[2]=(_Float16)a0.z; af[3]=(_Float16)a0.w;
    af[4]=(_Float16)a1.x; af[5]=(_Float16)a1.y; af[6]=(_Float16)a1.z; af[7]=(_Float16)a1.w;
    af[8]=(_Float16)a2.x; af[9]=(_Float16)a2.y; af[10]=(_Float16)a2.z; af[11]=(_Float16)a2.w;
    af[12]=(_Float16)a3.x; af[13]=(_Float16)a3.y; af[14]=(_Float16)a3.z; af[15]=(_Float16)a3.w;
    return af;
}

// ============================================================================
// Generic GEMM: C[M,Nn] = act(A[M,K] @ B[K,Nn] + bias)   (row-major f32 in/out)
// block = 256 threads = 8 waves; block tile 128(M) x 64(N); wave tile 16x64.
// B tile (32x64) staged per k-step through LDS (transposed, f16), shared by
// all 8 waves; prefetched one step ahead.  A loaded per lane as 4x float4.
// M%128==0, Nn%64==0, K%32==0 at all call sites.
// ============================================================================
__global__ __launch_bounds__(256) void gemm_kernel(
    const float* __restrict__ A, const float* __restrict__ B,
    const float* __restrict__ bias, float* __restrict__ C,
    int M, int Nn, int K, int act)
{
    __shared__ __align__(16) _Float16 Bt[64 * 32];

    const int tid  = threadIdx.x;
    const int wave = tid >> 5;
    const int lane = tid & 31;
    const int hf   = lane >> 4;
    const int lr   = lane & 15;

    const int row0 = blockIdx.x * 128 + wave * 16;
    const int col0 = blockIdx.y * 64;
    const float* Arow = A + (size_t)(row0 + lr) * K;

    // B-tile cooperative fill mapping: q in [0,512): k = q>>4, n4 = (q&15)*4
    const int q0 = tid, q1 = tid + 256;
    const int k0 = q0 >> 4, n40 = (q0 & 15) * 4;
    const int k1 = q1 >> 4, n41 = (q1 & 15) * 4;

    float4 rb0 = *(const float4*)&B[(size_t)k0 * Nn + col0 + n40];
    float4 rb1 = *(const float4*)&B[(size_t)k1 * Nn + col0 + n41];

    v8f acc[4] = {};

    for (int kb = 0; kb < K; kb += 32) {
        __syncthreads();
        Bt[(n40 + 0) * 32 + k0] = (_Float16)rb0.x;
        Bt[(n40 + 1) * 32 + k0] = (_Float16)rb0.y;
        Bt[(n40 + 2) * 32 + k0] = (_Float16)rb0.z;
        Bt[(n40 + 3) * 32 + k0] = (_Float16)rb0.w;
        Bt[(n41 + 0) * 32 + k1] = (_Float16)rb1.x;
        Bt[(n41 + 1) * 32 + k1] = (_Float16)rb1.y;
        Bt[(n41 + 2) * 32 + k1] = (_Float16)rb1.z;
        Bt[(n41 + 3) * 32 + k1] = (_Float16)rb1.w;
        __syncthreads();
        if (kb + 32 < K) {
            rb0 = *(const float4*)&B[(size_t)(kb + 32 + k0) * Nn + col0 + n40];
            rb1 = *(const float4*)&B[(size_t)(kb + 32 + k1) * Nn + col0 + n41];
        }
        float4 a0 = *(const float4*)&Arow[kb + 8 * hf];
        float4 a1 = *(const float4*)&Arow[kb + 8 * hf + 4];
        float4 a2 = *(const float4*)&Arow[kb + 16 + 8 * hf];
        float4 a3 = *(const float4*)&Arow[kb + 16 + 8 * hf + 4];
        v16h af = pack_a(a0, a1, a2, a3);
#pragma unroll
        for (int s = 0; s < 4; ++s) {
            int nloc = s * 16 + lr;
            v8h c0 = *(const v8h*)&Bt[nloc * 32 + 8 * hf];
            v8h c1 = *(const v8h*)&Bt[nloc * 32 + 16 + 8 * hf];
            acc[s] = __builtin_amdgcn_wmma_f32_16x16x32_f16(
                false, af, false, joinv(c0, c1), (short)0, acc[s], false, false);
        }
    }
#pragma unroll
    for (int s = 0; s < 4; ++s) {
        int n = col0 + s * 16 + lr;
        float bv = bias ? bias[n] : 0.f;
#pragma unroll
        for (int r = 0; r < 8; ++r) {
            int m = row0 + r + hf * 8;
            float v = acc[s][r] + bv;
            if (act == 1) v = v > 0.f ? v : 0.f;
            C[(size_t)m * Nn + n] = v;
        }
    }
}

// ============================================================================
// Encoder stage 1: h1 = relu(concat(emb[N,128], dem[N,1]) @ We1[129,256] + be1)
// K padded 129 -> 160 with zeros; last k-step (k=128..159) handled specially.
// ============================================================================
__global__ __launch_bounds__(256) void enc1_kernel(
    const float* __restrict__ emb, const float* __restrict__ dem,
    const float* __restrict__ We1, const float* __restrict__ be1,
    float* __restrict__ C)
{
    __shared__ __align__(16) _Float16 Bt[64 * 32];

    const int tid  = threadIdx.x;
    const int wave = tid >> 5;
    const int lane = tid & 31;
    const int hf   = lane >> 4;
    const int lr   = lane & 15;

    const int row0 = blockIdx.x * 128 + wave * 16;
    const int col0 = blockIdx.y * 64;
    const int row  = row0 + lr;
    const float* Arow = emb + (size_t)row * 128;

    const int q0 = tid, q1 = tid + 256;
    const int k0 = q0 >> 4, n40 = (q0 & 15) * 4;
    const int k1 = q1 >> 4, n41 = (q1 & 15) * 4;

    float4 zero4 = make_float4(0.f, 0.f, 0.f, 0.f);
    float4 rb0 = *(const float4*)&We1[(size_t)k0 * KD + col0 + n40];
    float4 rb1 = *(const float4*)&We1[(size_t)k1 * KD + col0 + n41];

    v8f acc[4] = {};

    for (int kb = 0; kb < 160; kb += 32) {
        __syncthreads();
        Bt[(n40 + 0) * 32 + k0] = (_Float16)rb0.x;
        Bt[(n40 + 1) * 32 + k0] = (_Float16)rb0.y;
        Bt[(n40 + 2) * 32 + k0] = (_Float16)rb0.z;
        Bt[(n40 + 3) * 32 + k0] = (_Float16)rb0.w;
        Bt[(n41 + 0) * 32 + k1] = (_Float16)rb1.x;
        Bt[(n41 + 1) * 32 + k1] = (_Float16)rb1.y;
        Bt[(n41 + 2) * 32 + k1] = (_Float16)rb1.z;
        Bt[(n41 + 3) * 32 + k1] = (_Float16)rb1.w;
        __syncthreads();
        if (kb + 32 < 160) {
            int kk0 = kb + 32 + k0, kk1 = kb + 32 + k1;
            rb0 = (kk0 < 129) ? *(const float4*)&We1[(size_t)kk0 * KD + col0 + n40] : zero4;
            rb1 = (kk1 < 129) ? *(const float4*)&We1[(size_t)kk1 * KD + col0 + n41] : zero4;
        }
        v16h af;
        if (kb < 128) {
            float4 a0 = *(const float4*)&Arow[kb + 8 * hf];
            float4 a1 = *(const float4*)&Arow[kb + 8 * hf + 4];
            float4 a2 = *(const float4*)&Arow[kb + 16 + 8 * hf];
            float4 a3 = *(const float4*)&Arow[kb + 16 + 8 * hf + 4];
            af = pack_a(a0, a1, a2, a3);
        } else {
            // k in [128,160): only k==128 is live (the demand column)
            float dv = dem[row];
#pragma unroll
            for (int i = 0; i < 16; ++i) af[i] = (_Float16)0.f;
            if (hf == 0) af[0] = (_Float16)dv;   // i=0,hf=0 -> k = kb+0 = 128
        }
#pragma unroll
        for (int s = 0; s < 4; ++s) {
            int nloc = s * 16 + lr;
            v8h c0 = *(const v8h*)&Bt[nloc * 32 + 8 * hf];
            v8h c1 = *(const v8h*)&Bt[nloc * 32 + 16 + 8 * hf];
            acc[s] = __builtin_amdgcn_wmma_f32_16x16x32_f16(
                false, af, false, joinv(c0, c1), (short)0, acc[s], false, false);
        }
    }
#pragma unroll
    for (int s = 0; s < 4; ++s) {
        int n = col0 + s * 16 + lr;
        float bv = be1[n];
#pragma unroll
        for (int r = 0; r < 8; ++r) {
            int m = row0 + r + hf * 8;
            float v = acc[s][r] + bv;
            C[(size_t)m * KD + n] = v > 0.f ? v : 0.f;
        }
    }
}

// ============================================================================
// Gated update: z = sigmoid(nxt@Wz + enc@Uz + bz); enc' = z*nxt + (1-z)*enc
// Both weight tiles staged through LDS; two A streams loaded as float4.
// ============================================================================
__global__ __launch_bounds__(256) void gate_kernel(
    const float* __restrict__ nxt, const float* __restrict__ enc,
    const float* __restrict__ Wz, const float* __restrict__ Uz,
    const float* __restrict__ bz, float* __restrict__ encnew)
{
    __shared__ __align__(16) _Float16 Wt[64 * 32];
    __shared__ __align__(16) _Float16 Ut[64 * 32];

    const int tid  = threadIdx.x;
    const int wave = tid >> 5;
    const int lane = tid & 31;
    const int hf   = lane >> 4;
    const int lr   = lane & 15;

    const int row0 = blockIdx.x * 128 + wave * 16;
    const int col0 = blockIdx.y * 64;
    const float* Anrow = nxt + (size_t)(row0 + lr) * KD;
    const float* Aerow = enc + (size_t)(row0 + lr) * KD;

    const int q0 = tid, q1 = tid + 256;
    const int k0 = q0 >> 4, n40 = (q0 & 15) * 4;
    const int k1 = q1 >> 4, n41 = (q1 & 15) * 4;

    float4 rw0 = *(const float4*)&Wz[(size_t)k0 * KD + col0 + n40];
    float4 rw1 = *(const float4*)&Wz[(size_t)k1 * KD + col0 + n41];
    float4 ru0 = *(const float4*)&Uz[(size_t)k0 * KD + col0 + n40];
    float4 ru1 = *(const float4*)&Uz[(size_t)k1 * KD + col0 + n41];

    v8f acc[4] = {};

    for (int kb = 0; kb < KD; kb += 32) {
        __syncthreads();
        Wt[(n40+0)*32+k0]=(_Float16)rw0.x; Wt[(n40+1)*32+k0]=(_Float16)rw0.y;
        Wt[(n40+2)*32+k0]=(_Float16)rw0.z; Wt[(n40+3)*32+k0]=(_Float16)rw0.w;
        Wt[(n41+0)*32+k1]=(_Float16)rw1.x; Wt[(n41+1)*32+k1]=(_Float16)rw1.y;
        Wt[(n41+2)*32+k1]=(_Float16)rw1.z; Wt[(n41+3)*32+k1]=(_Float16)rw1.w;
        Ut[(n40+0)*32+k0]=(_Float16)ru0.x; Ut[(n40+1)*32+k0]=(_Float16)ru0.y;
        Ut[(n40+2)*32+k0]=(_Float16)ru0.z; Ut[(n40+3)*32+k0]=(_Float16)ru0.w;
        Ut[(n41+0)*32+k1]=(_Float16)ru1.x; Ut[(n41+1)*32+k1]=(_Float16)ru1.y;
        Ut[(n41+2)*32+k1]=(_Float16)ru1.z; Ut[(n41+3)*32+k1]=(_Float16)ru1.w;
        __syncthreads();
        if (kb + 32 < KD) {
            rw0 = *(const float4*)&Wz[(size_t)(kb+32+k0)*KD + col0 + n40];
            rw1 = *(const float4*)&Wz[(size_t)(kb+32+k1)*KD + col0 + n41];
            ru0 = *(const float4*)&Uz[(size_t)(kb+32+k0)*KD + col0 + n40];
            ru1 = *(const float4*)&Uz[(size_t)(kb+32+k1)*KD + col0 + n41];
        }
        float4 n0 = *(const float4*)&Anrow[kb + 8*hf];
        float4 n1 = *(const float4*)&Anrow[kb + 8*hf + 4];
        float4 n2 = *(const float4*)&Anrow[kb + 16 + 8*hf];
        float4 n3 = *(const float4*)&Anrow[kb + 16 + 8*hf + 4];
        v16h af1 = pack_a(n0, n1, n2, n3);
        float4 e0 = *(const float4*)&Aerow[kb + 8*hf];
        float4 e1 = *(const float4*)&Aerow[kb + 8*hf + 4];
        float4 e2 = *(const float4*)&Aerow[kb + 16 + 8*hf];
        float4 e3 = *(const float4*)&Aerow[kb + 16 + 8*hf + 4];
        v16h af2 = pack_a(e0, e1, e2, e3);
#pragma unroll
        for (int s = 0; s < 4; ++s) {
            int nloc = s * 16 + lr;
            v8h w0 = *(const v8h*)&Wt[nloc * 32 + 8 * hf];
            v8h w1 = *(const v8h*)&Wt[nloc * 32 + 16 + 8 * hf];
            acc[s] = __builtin_amdgcn_wmma_f32_16x16x32_f16(
                false, af1, false, joinv(w0, w1), (short)0, acc[s], false, false);
            v8h u0 = *(const v8h*)&Ut[nloc * 32 + 8 * hf];
            v8h u1 = *(const v8h*)&Ut[nloc * 32 + 16 + 8 * hf];
            acc[s] = __builtin_amdgcn_wmma_f32_16x16x32_f16(
                false, af2, false, joinv(u0, u1), (short)0, acc[s], false, false);
        }
    }
#pragma unroll
    for (int s = 0; s < 4; ++s) {
        int n = col0 + s * 16 + lr;
        float bv = bz[n];
#pragma unroll
        for (int r = 0; r < 8; ++r) {
            int m = row0 + r + hf * 8;
            float z  = sigmoidf_(acc[s][r] + bv);
            float nv = nxt[(size_t)m * KD + n];
            float ev = enc[(size_t)m * KD + n];
            encnew[(size_t)m * KD + n] = z * nv + (1.f - z) * ev;
        }
    }
}

// ============================================================================
// s1/s2 per (head,row): wave-per-row dot of t[h,n,:] with a1[h,:], a2[h,:]
// ============================================================================
__global__ __launch_bounds__(256) void s_kernel(
    const float* __restrict__ t, const float* __restrict__ a1,
    const float* __restrict__ a2, float* __restrict__ s1, float* __restrict__ s2)
{
    const int wave = threadIdx.x >> 5;
    const int lane = threadIdx.x & 31;
    const int rid  = blockIdx.x * 8 + wave;       // 0 .. H*N-1
    const int h    = rid / KN;
    const int n    = rid - h * KN;

    const float* tr = t + (size_t)h * KN * KD + (size_t)n * KD;
    float acc1 = 0.f, acc2 = 0.f;
    for (int d = lane; d < KD; d += 32) {
        float tv = tr[d];
        acc1 += tv * a1[h * KD + d];
        acc2 += tv * a2[h * KD + d];
    }
#pragma unroll
    for (int off = 16; off > 0; off >>= 1) {
        acc1 += __shfl_xor(acc1, off, 32);
        acc2 += __shfl_xor(acc2, off, 32);
    }
    if (lane == 0) {
        s1[h * KN + n] = acc1;
        s2[h * KN + n] = acc2;
    }
}

// ============================================================================
// Masked-softmax row stats: e = lrelu(s1[n]+s2[m]) over adj[n,m]!=0; grid (N,H)
// ============================================================================
__global__ __launch_bounds__(256) void rowstats_kernel(
    const float* __restrict__ adj, const float* __restrict__ s1,
    const float* __restrict__ s2, float* __restrict__ rmax, float* __restrict__ rsum)
{
    const int n = blockIdx.x, h = blockIdx.y, tid = threadIdx.x;
    const float s1v = s1[h * KN + n];
    const float* ar = adj + (size_t)n * KN;
    const float* s2r = s2 + h * KN;

    float mx = -1e30f, sm = 0.f;
    for (int m = tid; m < KN; m += 256) {
        if (ar[m] != 0.f) {
            float e = lrelu02(s1v + s2r[m]);
            if (e > mx) { sm = sm * __expf(mx - e) + 1.f; mx = e; }
            else        { sm += __expf(e - mx); }
        }
    }
    __shared__ float smax[256], ssum[256];
    smax[tid] = mx; ssum[tid] = sm;
    __syncthreads();
    for (int off = 128; off > 0; off >>= 1) {
        if (tid < off) {
            float m2 = smax[tid + off], s2v = ssum[tid + off];
            float M = fmaxf(smax[tid], m2);
            ssum[tid] = ssum[tid] * __expf(smax[tid] - M) + s2v * __expf(m2 - M);
            smax[tid] = M;
        }
        __syncthreads();
    }
    if (tid == 0) { rmax[h * KN + n] = smax[0]; rsum[h * KN + n] = ssum[0]; }
}

// ============================================================================
// Flash-style attention GEMM: hout[h] = softmax_masked(e) @ t[h]
// A-fragment (attention weights) computed on the fly; t chunk (32xD) staged
// through LDS transposed in f16 (shared by all 8 waves), prefetched.
// grid = (N/32, H); block 8 waves: 2 row-groups x 4 col-groups of 64.
// ============================================================================
__global__ __launch_bounds__(256) void attn_kernel(
    const float* __restrict__ adj, const float* __restrict__ s1,
    const float* __restrict__ s2, const float* __restrict__ rmax,
    const float* __restrict__ rsum, const float* __restrict__ t,
    float* __restrict__ hout)
{
    __shared__ __align__(16) _Float16 Bt[KD * 32];   // t-chunk transposed: [n][k]

    const int tid  = threadIdx.x;
    const int wave = tid >> 5;
    const int lane = tid & 31;
    const int hf   = lane >> 4;
    const int lr   = lane & 15;
    const int h    = blockIdx.y;

    const int row0 = blockIdx.x * 32 + (wave >> 2) * 16;
    const int col0 = (wave & 3) * 64;
    const int row  = row0 + lr;                    // A-row owned by this lane

    const float s1v = s1[h * KN + row];
    const float mx  = rmax[h * KN + row];
    const float inv = 1.f / rsum[h * KN + row];
    const float* ar  = adj + (size_t)row * KN;
    const float* s2r = s2 + h * KN;
    const float* th  = t + (size_t)h * KN * KD;

    // cooperative fill: 32 x 256 floats = 2048 float4; q = k*64 + nq
    int kq[8], nq4[8];
    float4 rb[8];
#pragma unroll
    for (int p = 0; p < 8; ++p) {
        int q = tid + p * 256;
        kq[p] = q >> 6; nq4[p] = (q & 63) * 4;
        rb[p] = *(const float4*)&th[(size_t)kq[p] * KD + nq4[p]];
    }

    v8f acc[4] = {};

    for (int kb = 0; kb < KN; kb += 32) {
        __syncthreads();
#pragma unroll
        for (int p = 0; p < 8; ++p) {
            Bt[(nq4[p] + 0) * 32 + kq[p]] = (_Float16)rb[p].x;
            Bt[(nq4[p] + 1) * 32 + kq[p]] = (_Float16)rb[p].y;
            Bt[(nq4[p] + 2) * 32 + kq[p]] = (_Float16)rb[p].z;
            Bt[(nq4[p] + 3) * 32 + kq[p]] = (_Float16)rb[p].w;
        }
        __syncthreads();
        if (kb + 32 < KN) {
#pragma unroll
            for (int p = 0; p < 8; ++p)
                rb[p] = *(const float4*)&th[(size_t)(kb + 32 + kq[p]) * KD + nq4[p]];
        }
        // A-fragment: masked softmax weights for this lane's row, m-chunk
        float4 aj0 = *(const float4*)&ar[kb + 8 * hf];
        float4 aj1 = *(const float4*)&ar[kb + 8 * hf + 4];
        float4 aj2 = *(const float4*)&ar[kb + 16 + 8 * hf];
        float4 aj3 = *(const float4*)&ar[kb + 16 + 8 * hf + 4];
        float4 sb0 = *(const float4*)&s2r[kb + 8 * hf];
        float4 sb1 = *(const float4*)&s2r[kb + 8 * hf + 4];
        float4 sb2 = *(const float4*)&s2r[kb + 16 + 8 * hf];
        float4 sb3 = *(const float4*)&s2r[kb + 16 + 8 * hf + 4];
        float ajv[16] = {aj0.x,aj0.y,aj0.z,aj0.w, aj1.x,aj1.y,aj1.z,aj1.w,
                         aj2.x,aj2.y,aj2.z,aj2.w, aj3.x,aj3.y,aj3.z,aj3.w};
        float sbv[16] = {sb0.x,sb0.y,sb0.z,sb0.w, sb1.x,sb1.y,sb1.z,sb1.w,
                         sb2.x,sb2.y,sb2.z,sb2.w, sb3.x,sb3.y,sb3.z,sb3.w};
        v16h af;
#pragma unroll
        for (int i = 0; i < 16; ++i) {
            float w = 0.f;
            if (ajv[i] != 0.f) {
                float e = lrelu02(s1v + sbv[i]);
                w = __expf(e - mx) * inv;
            }
            af[i] = (_Float16)w;
        }
#pragma unroll
        for (int s = 0; s < 4; ++s) {
            int nloc = col0 + s * 16 + lr;
            v8h c0 = *(const v8h*)&Bt[nloc * 32 + 8 * hf];
            v8h c1 = *(const v8h*)&Bt[nloc * 32 + 16 + 8 * hf];
            acc[s] = __builtin_amdgcn_wmma_f32_16x16x32_f16(
                false, af, false, joinv(c0, c1), (short)0, acc[s], false, false);
        }
    }
    float* out = hout + (size_t)h * KN * KD;
#pragma unroll
    for (int s = 0; s < 4; ++s) {
        int n = col0 + s * 16 + lr;
#pragma unroll
        for (int r = 0; r < 8; ++r) {
            int m = row0 + r + hf * 8;
            out[(size_t)m * KD + n] = acc[s][r];
        }
    }
}

// mean over heads: gat[n,d] = (1/8) * sum_h hout[h,n,d]
__global__ __launch_bounds__(256) void mean_kernel(
    const float* __restrict__ hout, float* __restrict__ gat)
{
    const size_t idx = (size_t)blockIdx.x * 256 + threadIdx.x;  // < N*D
    float acc = 0.f;
#pragma unroll
    for (int h = 0; h < KH; ++h) acc += hout[(size_t)h * KN * KD + idx];
    gat[idx] = acc * 0.125f;
}

// Masked row softmax in place: pred[n,:] -> props[n,:] (masked entries -> 0)
__global__ __launch_bounds__(256) void props_kernel(
    const float* __restrict__ adj, float* __restrict__ pred)
{
    const int n = blockIdx.x, tid = threadIdx.x;
    const float* ar = adj + (size_t)n * KN;
    float* pr = pred + (size_t)n * KN;

    float mx = -1e30f, sm = 0.f;
    for (int m = tid; m < KN; m += 256) {
        if (ar[m] != 0.f) {
            float e = pr[m];
            if (e > mx) { sm = sm * __expf(mx - e) + 1.f; mx = e; }
            else        { sm += __expf(e - mx); }
        }
    }
    __shared__ float smax[256], ssum[256];
    smax[tid] = mx; ssum[tid] = sm;
    __syncthreads();
    for (int off = 128; off > 0; off >>= 1) {
        if (tid < off) {
            float m2 = smax[tid + off], s2v = ssum[tid + off];
            float M = fmaxf(smax[tid], m2);
            ssum[tid] = ssum[tid] * __expf(smax[tid] - M) + s2v * __expf(m2 - M);
            smax[tid] = M;
        }
        __syncthreads();
    }
    const float MX = smax[0], INV = 1.f / ssum[0];
    __syncthreads();
    for (int m = tid; m < KN; m += 256)
        pr[m] = (ar[m] != 0.f) ? __expf(pr[m] - MX) * INV : 0.f;
}

// q[i] = sum_j props[i,j]^2
__global__ __launch_bounds__(256) void qrow_kernel(
    const float* __restrict__ props, float* __restrict__ q)
{
    const int i = blockIdx.x, tid = threadIdx.x;
    const float* pr = props + (size_t)i * KN;
    float acc = 0.f;
    for (int j = tid; j < KN; j += 256) { float v = pr[j]; acc += v * v; }
    __shared__ float sh[256];
    sh[tid] = acc; __syncthreads();
    for (int off = 128; off > 0; off >>= 1) {
        if (tid < off) sh[tid] += sh[tid + off];
        __syncthreads();
    }
    if (tid == 0) q[i] = sh[0];
}

// r[i] = relu((haveU ? u[i] : 0) - dem[i])
__global__ __launch_bounds__(256) void r_kernel(
    const float* __restrict__ u, const float* __restrict__ dem,
    float* __restrict__ r, int haveU)
{
    const int i = blockIdx.x * 256 + threadIdx.x;
    float v = (haveU ? u[i] : 0.f) - dem[i];
    r[i] = v > 0.f ? v : 0.f;
}

// y[j] = sum_i props[i,j] * r[i]   (coalesced across j; r tiled through LDS)
__global__ __launch_bounds__(256) void mv_kernel(
    const float* __restrict__ P, const float* __restrict__ r, float* __restrict__ y)
{
    const int j = blockIdx.x * 256 + threadIdx.x;
    __shared__ float rs[256];
    float acc = 0.f;
    for (int i0 = 0; i0 < KN; i0 += 256) {
        rs[threadIdx.x] = r[i0 + threadIdx.x];
        __syncthreads();
        for (int ii = 0; ii < 256; ++ii)
            acc += P[(size_t)(i0 + ii) * KN + j] * rs[ii];
        __syncthreads();
    }
    y[j] = acc;
}

// dual vars: v[n] = sum_d d1[n,d]*Wv2[d] + bv2  (wave per row)
__global__ __launch_bounds__(256) void v_kernel(
    const float* __restrict__ d1, const float* __restrict__ Wv2,
    const float* __restrict__ bv2, float* __restrict__ v)
{
    const int wave = threadIdx.x >> 5;
    const int lane = threadIdx.x & 31;
    const int n = blockIdx.x * 8 + wave;
    float acc = 0.f;
    for (int d = lane; d < KD; d += 32) acc += d1[(size_t)n * KD + d] * Wv2[d];
#pragma unroll
    for (int off = 16; off > 0; off >>= 1) acc += __shfl_xor(acc, off, 32);
    if (lane == 0) v[n] = acc + bv2[0];
}

// partial A: pa[b] = sum r[i]^2*q[i];  pb[b] = sum v[i]*dem[i]   (grid 12)
__global__ __launch_bounds__(256) void partialA_kernel(
    const float* __restrict__ r, const float* __restrict__ q,
    const float* __restrict__ v, const float* __restrict__ dem,
    float* __restrict__ pa, float* __restrict__ pb)
{
    const int i = blockIdx.x * 256 + threadIdx.x;
    __shared__ float sa[256], sb[256];
    sa[threadIdx.x] = r[i] * r[i] * q[i];
    sb[threadIdx.x] = v[i] * dem[i];
    __syncthreads();
    for (int off = 128; off > 0; off >>= 1) {
        if (threadIdx.x < off) { sa[threadIdx.x] += sa[threadIdx.x + off];
                                 sb[threadIdx.x] += sb[threadIdx.x + off]; }
        __syncthreads();
    }
    if (threadIdx.x == 0) { pa[blockIdx.x] = sa[0]; pb[blockIdx.x] = sb[0]; }
}

// partial D: dual flow cost terms over [N,N]: adj && dd>0 -> -dd^2/4  (grid 192)
__global__ __launch_bounds__(256) void partialD_kernel(
    const float* __restrict__ adj, const float* __restrict__ v, float* __restrict__ pd)
{
    const size_t total = (size_t)KN * KN;
    const size_t stride = (size_t)192 * 256;
    float acc = 0.f;
    for (size_t idx = (size_t)blockIdx.x * 256 + threadIdx.x; idx < total; idx += stride) {
        if (adj[idx] != 0.f) {
            int i = (int)(idx / KN), j = (int)(idx - (size_t)i * KN);
            float dd = v[i] - v[j];
            if (dd > 0.f) acc -= 0.25f * dd * dd;
        }
    }
    __shared__ float sh[256];
    sh[threadIdx.x] = acc; __syncthreads();
    for (int off = 128; off > 0; off >>= 1) {
        if (threadIdx.x < off) sh[threadIdx.x] += sh[threadIdx.x + off];
        __syncthreads();
    }
    if (threadIdx.x == 0) pd[blockIdx.x] = sh[0];
}

// out = flow_cost - dual_cost = sumA - sumD + sumB   (serial, deterministic)
__global__ void finalize_kernel(
    const float* __restrict__ pa, const float* __restrict__ pb,
    const float* __restrict__ pd, float* __restrict__ out)
{
    float A = 0.f, B = 0.f, D = 0.f;
    for (int i = 0; i < 12; ++i)  { A += pa[i]; B += pb[i]; }
    for (int i = 0; i < 192; ++i) { D += pd[i]; }
    out[0] = A - D + B;
}

// ============================================================================
extern "C" void kernel_launch(void* const* d_in, const int* in_sizes, int n_in,
                              void* d_out, int out_size, void* d_ws, size_t ws_size,
                              hipStream_t stream) {
    (void)in_sizes; (void)n_in; (void)out_size; (void)ws_size;
    const float* emb  = (const float*)d_in[0];
    const float* dem  = (const float*)d_in[1];
    const float* adj  = (const float*)d_in[2];
    const float* We1  = (const float*)d_in[3];
    const float* be1  = (const float*)d_in[4];
    const float* We2  = (const float*)d_in[5];
    const float* be2  = (const float*)d_in[6];
    const float* Wg   = (const float*)d_in[7];
    const float* bg   = (const float*)d_in[8];
    const float* a1   = (const float*)d_in[9];
    const float* a2   = (const float*)d_in[10];
    const float* Wz   = (const float*)d_in[11];
    const float* Uz   = (const float*)d_in[12];
    const float* bz   = (const float*)d_in[13];
    const float* Wd1  = (const float*)d_in[14];
    const float* bd1  = (const float*)d_in[15];
    const float* Wd2  = (const float*)d_in[16];
    const float* bd2  = (const float*)d_in[17];
    const float* Wv1  = (const float*)d_in[18];
    const float* bv1  = (const float*)d_in[19];
    const float* Wv2  = (const float*)d_in[20];
    const float* bv2  = (const float*)d_in[21];
    float* out = (float*)d_out;

    // ---- workspace layout (aligned 256B) ----
    char* w = (char*)d_ws;
    size_t off = 0;
    auto alloc = [&](size_t bytes) -> float* {
        float* p = (float*)(w + off);
        off += (bytes + 255) & ~(size_t)255;
        return p;
    };
    const size_t NDb = (size_t)KN * KD * 4;
    float* enc_a = alloc(NDb);
    float* enc_b = alloc(NDb);
    float* h1    = alloc(NDb);                 // also d1
    float* gat   = alloc(NDb);
    float* s1b   = alloc((size_t)KH * KN * 4);
    float* s2b   = alloc((size_t)KH * KN * 4);
    float* rmaxb = alloc((size_t)KH * KN * 4);
    float* rsumb = alloc((size_t)KH * KN * 4);
    float* ubuf  = alloc((size_t)KN * 4);
    float* rbuf  = alloc((size_t)KN * 4);
    float* qbuf  = alloc((size_t)KN * 4);
    float* vbuf  = alloc((size_t)KN * 4);
    float* pa    = alloc(12 * 4);
    float* pb    = alloc(12 * 4);
    float* pd    = alloc(192 * 4);
    float* big   = alloc(2 * (size_t)KH * KN * KD * 4); // t + hout; reused as pred/props
    float* tbuf  = big;
    float* hout  = big + (size_t)KH * KN * KD;
    float* pred  = big;                                  // reuse after GAT layers

    dim3 blk(256);
    dim3 g_small(KN / 128, KD / 64);     // 24 x 4
    dim3 g_pred(KN / 128, KN / 64);      // 24 x 48
    dim3 g_rows(KN, KH);                 // 3072 x 8
    dim3 g_attn(KN / 32, KH);            // 96 x 8

    // --- node encoder MLP ---
    enc1_kernel<<<g_small, blk, 0, stream>>>(emb, dem, We1, be1, h1);
    gemm_kernel<<<g_small, blk, 0, stream>>>(h1, We2, be2, enc_a, KN, KD, KD, 1);

    // --- GAT layers with gated update ---
    float* encp = enc_a;
    float* encn = enc_b;
    for (int layer = 0; layer < 2; ++layer) {
        for (int h = 0; h < KH; ++h)
            gemm_kernel<<<g_small, blk, 0, stream>>>(
                encp, Wg + (size_t)h * KD * KD, bg + (size_t)h * KD,
                tbuf + (size_t)h * KN * KD, KN, KD, KD, 0);
        s_kernel<<<dim3(KH * KN / 8), blk, 0, stream>>>(tbuf, a1, a2, s1b, s2b);
        rowstats_kernel<<<g_rows, blk, 0, stream>>>(adj, s1b, s2b, rmaxb, rsumb);
        attn_kernel<<<g_attn, blk, 0, stream>>>(adj, s1b, s2b, rmaxb, rsumb, tbuf, hout);
        mean_kernel<<<dim3((KN * KD) / 256), blk, 0, stream>>>(hout, gat);
        gate_kernel<<<g_small, blk, 0, stream>>>(gat, encp, Wz, Uz, bz, encn);
        float* tmp = encp; encp = encn; encn = tmp;
    }

    // --- decoder -> props ---  (t/hout dead now; pred overlaps that region)
    gemm_kernel<<<g_small, blk, 0, stream>>>(encp, Wd1, bd1, h1, KN, KD, KD, 1);
    gemm_kernel<<<g_pred, blk, 0, stream>>>(h1, Wd2, bd2, pred, KN, KN, KD, 0);
    props_kernel<<<dim3(KN), blk, 0, stream>>>(adj, pred);

    // --- flow iterations (vectorized scan): u_{k+1} = P^T relu(u_k - d) ---
    qrow_kernel<<<dim3(KN), blk, 0, stream>>>(pred, qbuf);
    r_kernel<<<dim3(KN / 256), blk, 0, stream>>>(ubuf, dem, rbuf, 0);  // r_0 = relu(-d)
    for (int it = 0; it < 49; ++it) {
        mv_kernel<<<dim3(KN / 256), blk, 0, stream>>>(pred, rbuf, ubuf);
        r_kernel<<<dim3(KN / 256), blk, 0, stream>>>(ubuf, dem, rbuf, 1);
    }
    // rbuf == r_49; flow_cost = sum_i r_49[i]^2 * q[i]

    // --- dual variables ---
    gemm_kernel<<<g_small, blk, 0, stream>>>(encp, Wv1, bv1, h1, KN, KD, KD, 1);
    v_kernel<<<dim3(KN / 8), blk, 0, stream>>>(h1, Wv2, bv2, vbuf);

    // --- reductions + final scalar ---
    partialA_kernel<<<dim3(KN / 256), blk, 0, stream>>>(rbuf, qbuf, vbuf, dem, pa, pb);
    partialD_kernel<<<dim3(192), blk, 0, stream>>>(adj, vbuf, pd);
    finalize_kernel<<<dim3(1), dim3(1), 0, stream>>>(pa, pb, pd, out);
}
